// OriginalVanillaSFNN_86474871538228
// MI455X (gfx1250) — compile-verified, matrix operand order checked by reading, not run on hardware
//
#include <hip/hip_runtime.h>

typedef float    v2f  __attribute__((ext_vector_type(2)));
typedef float    v8f  __attribute__((ext_vector_type(8)));
typedef _Float16 v16h __attribute__((ext_vector_type(16)));

#define BB   256
#define TT   250
#define DD   700
#define HH   1024
#define OO   20
#define APAD 708   // padded row stride (floats) for the staged input tile

// CDNA5 async copy: 16B global -> LDS, tracked on ASYNCcnt (ISA §15.18.3 op 98).
__device__ __forceinline__ void async_copy_b128(unsigned lds_byte_off,
                                                const float* gptr) {
  asm volatile("global_load_async_to_lds_b128 %0, %1, off"
               :: "v"(lds_byte_off),
                  "v"((unsigned long long)(uintptr_t)gptr)
               : "memory");
}
__device__ __forceinline__ void wait_async0() {
  asm volatile("s_wait_asynccnt 0x0" ::: "memory");
}

// ---------------------------------------------------------------------------
// Kernel 1: fused  d1 = input @ W1^T + b1  and LIF layer-1 recurrence.
// grid = (H/128, B/16); block = 256 (8 waves). Wave w owns h-tile [h0, h0+16).
// Double-buffered A tile in LDS, prefetched with global_load_async_to_lds_b128.
// State mem1/spk1 lives in the WMMA C layout (8 VGPRs each) across all T.
// Spikes emitted bit-packed: u16 per (t, b, h/16).
// ---------------------------------------------------------------------------
__global__ __launch_bounds__(256) void snn_layer1(
    const float* __restrict__ in,   // [B, T, D]
    const float* __restrict__ W1,   // [H, D]
    const float* __restrict__ b1,   // [H]
    const float* __restrict__ tau1, // [H]
    unsigned short* __restrict__ spkout) // [T][B][H/16] u16
{
  __shared__ float aLDS[2][16 * APAD];  // 2 x 45.3 KB (CDNA5 WGP LDS = 320 KB)

  const int tid   = threadIdx.x;
  const int lane  = tid & 31;
  const int wv    = tid >> 5;      // 0..7
  const int l16   = lane & 15;     // m for A-frag, n for B/C-frag
  const int khalf = lane >> 4;     // 0/1
  const int b0    = blockIdx.y * 16;
  const int h0    = blockIdx.x * 128 + wv * 16;
  const int h16   = h0 >> 4;

  const int   hcol  = h0 + l16;
  const float bias  = b1[hcol];
  const float alpha = 1.0f / (1.0f + __expf(-tau1[hcol]));
  const float oma   = 1.0f - alpha;
  const float* __restrict__ w1row = W1 + (size_t)hcol * DD; // B[k][n] = W1[h0+n][k]

  // Async-stage input tile [16 x 700] for timestep t into buffer pb:
  // wave wv copies batch rows 2wv, 2wv+1; 175 aligned 16-B chunks per row.
  auto stage = [&](int t, int pb) {
    const float* src0 = in + ((size_t)(b0 + 2 * wv)     * TT + t) * DD;
    const float* src1 = in + ((size_t)(b0 + 2 * wv + 1) * TT + t) * DD;
    unsigned ld0 = (unsigned)(uintptr_t)&aLDS[pb][(2 * wv)     * APAD];
    unsigned ld1 = (unsigned)(uintptr_t)&aLDS[pb][(2 * wv + 1) * APAD];
    for (int c = lane; c < DD / 4; c += 32) {
      async_copy_b128(ld0 + (unsigned)c * 16u, src0 + c * 4);
      async_copy_b128(ld1 + (unsigned)c * 16u, src1 + c * 4);
    }
  };

  v8f mem = {};
  v8f spk = {};

  stage(0, 0);
  wait_async0();
  __syncthreads();

  for (int t = 0; t < TT; ++t) {
    const int pb = t & 1;
    if (t + 1 < TT) stage(t + 1, pb ^ 1);   // prefetch overlaps compute

    // GEMM tile: C[16b x 16h] += A[16 x 700] * B[700 x 16], K-step 4 (f32 WMMA)
    v8f c = {};
    const float* arow = &aLDS[pb][l16 * APAD + 2 * khalf];
    for (int k = 0; k < DD; k += 4) {
      v2f a = *(const v2f*)(arow + k);                 // A[m][k+2kh], +1
      v2f b = *(const v2f*)(w1row + k + 2 * khalf);    // B[k+2kh][n], +1
      c = __builtin_amdgcn_wmma_f32_16x16x4_f32(false, a, false, b,
                                                (short)0, c, false, false);
    }

    // LIF update in C layout: lane holds n=l16, VGPR r holds m = r + 8*khalf.
    unsigned msk[8];
#pragma unroll
    for (int r = 0; r < 8; ++r) {
      float d1 = c[r] + bias;
      float m1 = mem[r] * alpha + oma * d1 - spk[r];   // VTH=1, DT=1, soft reset
      mem[r] = m1;
      bool sp = (m1 - 1.0f) > 0.0f;
      spk[r] = sp ? 1.0f : 0.0f;
      // ballot bit L <-> (m = r + 8*(L/16), n = L%16); result is wave-uniform
      msk[r] = __builtin_amdgcn_ballot_w32(sp);
    }
    // Lane L (<16) stores the u16 spike word of batch row m=L:
    //   rows 0..7 -> msk[m][15:0], rows 8..15 -> msk[m-8][31:16].
    if (lane < 16) {
      unsigned sel = msk[0];
#pragma unroll
      for (int r = 1; r < 8; ++r) sel = ((lane & 7) == r) ? msk[r] : sel;
      unsigned val = (lane >> 3) ? (sel >> 16) : (sel & 0xFFFFu);
      spkout[((size_t)t * BB + (b0 + lane)) * (HH / 16) + h16] =
          (unsigned short)val;
    }

    wait_async0();     // own prefetch complete (no-op on last iteration)
    __syncthreads();   // all waves' prefetch complete; tile t fully consumed
  }
}

// ---------------------------------------------------------------------------
// Kernel 2: layer-2 readout + softmax accumulation, sequential over T.
// grid = B/16; block = 256 (8 waves). Wave w reduces K-chunk [w*128,(w+1)*128).
// Spikes expanded bit->f16 into WMMA A-fragments; W2 fragments preloaded.
// ---------------------------------------------------------------------------
__global__ __launch_bounds__(256) void snn_layer2(
    const unsigned short* __restrict__ spk, // [T][B][H/16]
    const float* __restrict__ W2,           // [O, H]
    const float* __restrict__ b2,           // [O]
    const float* __restrict__ tau2,         // [O]
    float* __restrict__ out)                // [B, O]
{
  __shared__ float buf[8][16][32];  // per-wave partial d2 tiles
  __shared__ float mem2s[16][OO];
  __shared__ float accs[16][OO];
  __shared__ float a2s[OO];
  __shared__ float b2s[OO];

  const int tid   = threadIdx.x;
  const int lane  = tid & 31;
  const int wv    = tid >> 5;
  const int l16   = lane & 15;
  const int khalf = lane >> 4;
  const int b0    = blockIdx.x * 16;

  if (tid < OO) {
    a2s[tid] = 1.0f / (1.0f + __expf(-tau2[tid]));
    b2s[tid] = b2[tid];
  }
  for (int i = tid; i < 16 * OO; i += 256) {
    mem2s[i / OO][i % OO] = 0.0f;
    accs[i / OO][i % OO]  = 0.0f;
  }

  // Preload W2 B-fragments (t-invariant): 2 n-tiles x 4 k-steps of 32.
  v16h Bf[2][4];
  for (int nt = 0; nt < 2; ++nt) {
    const int oo = nt * 16 + l16;
    for (int kt = 0; kt < 4; ++kt) {
      const int kbase = wv * 128 + kt * 32;
      v16h f;
#pragma unroll
      for (int j = 0; j < 8; ++j) {
        const int kk = kbase + ((j < 4) ? 0 : 16) + 8 * khalf + 2 * (j & 3);
        float x0 = 0.0f, x1 = 0.0f;
        if (oo < OO) {
          x0 = W2[(size_t)oo * HH + kk];
          x1 = W2[(size_t)oo * HH + kk + 1];
        }
        f[2 * j]     = (_Float16)x0;
        f[2 * j + 1] = (_Float16)x1;
      }
      Bf[nt][kt] = f;
    }
  }
  __syncthreads();

  const unsigned* __restrict__ spk32 = (const unsigned*)spk;

  for (int t = 0; t < TT; ++t) {
    v8f c0 = {}, c1 = {};
#pragma unroll
    for (int kt = 0; kt < 4; ++kt) {
      const int kbase = wv * 128 + kt * 32;
      const unsigned w32 =
          spk32[((size_t)t * BB + (b0 + l16)) * (HH / 32) + (kbase >> 5)];
      v16h a;
#pragma unroll
      for (int j = 0; j < 8; ++j) {
        const int kk = ((j < 4) ? 0 : 16) + 8 * khalf + 2 * (j & 3);
        a[2 * j]     = ((w32 >> kk) & 1u)       ? (_Float16)1.0f : (_Float16)0.0f;
        a[2 * j + 1] = ((w32 >> (kk + 1)) & 1u) ? (_Float16)1.0f : (_Float16)0.0f;
      }
      c0 = __builtin_amdgcn_wmma_f32_16x16x32_f16(false, a, false, Bf[0][kt],
                                                  (short)0, c0, false, false);
      c1 = __builtin_amdgcn_wmma_f32_16x16x32_f16(false, a, false, Bf[1][kt],
                                                  (short)0, c1, false, false);
    }
    // Dump per-wave partial C tiles.
#pragma unroll
    for (int r = 0; r < 8; ++r) {
      const int mrow = r + 8 * khalf;
      buf[wv][mrow][l16]      = c0[r];
      buf[wv][mrow][16 + l16] = c1[r];
    }
    __syncthreads();

    // Sequential state update: thread b handles one batch row.
    if (tid < 16) {
      float mv[OO];
      float mx = -1e30f;
      for (int o = 0; o < OO; ++o) {
        float d2 = b2s[o];
        for (int w = 0; w < 8; ++w) d2 += buf[w][tid][o];
        const float al = a2s[o];
        const float m2 = mem2s[tid][o] * al + (1.0f - al) * d2;
        mem2s[tid][o] = m2;
        mv[o] = m2;
        mx = fmaxf(mx, m2);
      }
      if (t > 10) {
        float s = 0.0f;
        for (int o = 0; o < OO; ++o) { mv[o] = __expf(mv[o] - mx); s += mv[o]; }
        const float inv = 1.0f / s;
        for (int o = 0; o < OO; ++o) accs[tid][o] += mv[o] * inv;
      }
    }
    __syncthreads();
  }

  for (int i = tid; i < 16 * OO; i += 256)
    out[(size_t)(b0 + i / OO) * OO + (i % OO)] = accs[i / OO][i % OO];
}

// ---------------------------------------------------------------------------
extern "C" void kernel_launch(void* const* d_in, const int* in_sizes, int n_in,
                              void* d_out, int out_size, void* d_ws, size_t ws_size,
                              hipStream_t stream) {
  const float* in   = (const float*)d_in[0];
  const float* W1   = (const float*)d_in[1];
  const float* b1   = (const float*)d_in[2];
  const float* tau1 = (const float*)d_in[3];
  const float* W2   = (const float*)d_in[4];
  const float* b2   = (const float*)d_in[5];
  const float* tau2 = (const float*)d_in[6];

  // Spike bitmask workspace: [T][B][H/16] u16 = 8 MB (L2-resident).
  unsigned short* spk = (unsigned short*)d_ws;

  dim3 g1(HH / 128, BB / 16, 1);
  snn_layer1<<<g1, dim3(256), 0, stream>>>(in, W1, b1, tau1, spk);
  snn_layer2<<<dim3(BB / 16), dim3(256), 0, stream>>>(spk, W2, b2, tau2,
                                                      (float*)d_out);
}
